// MoE_48653389529538
// MI455X (gfx1250) — matrix operand order repaired and checked by reading, not compiled, not run on hardware
//
#include <hip/hip_runtime.h>
#include <hip/hip_bf16.h>

// ---------------- problem constants ----------------
#define T_TOK 4096
#define D_DIM 768
#define F_DIM 3072
#define NEXP  8

#define BM 128
#define BN 128
#define KS 32
#define LDA 40   // A tile row stride (bf16 elems); 80B, 16B-aligned
#define LDB 40   // B tile col stride (bf16 elems); 80B, 16B-aligned

typedef __attribute__((ext_vector_type(16))) __bf16 v16bf;
typedef __attribute__((ext_vector_type(8)))  __bf16 bf16x8;
typedef __attribute__((ext_vector_type(8)))  float  v8f;
typedef int v4i __attribute__((vector_size(16)));

#if defined(__HIPCC__) && __has_builtin(__builtin_amdgcn_global_load_async_to_lds_b128)
#define HAVE_ASYNC 1
#else
#define HAVE_ASYNC 0
#endif

// round f32 so that bits[31:16] are the RNE bf16 value
__device__ __forceinline__ unsigned rne_hi(float f) {
  unsigned u = __builtin_bit_cast(unsigned, f);
  return u + (0x7FFFu + ((u >> 16) & 1u));
}
__device__ __forceinline__ unsigned pack_bf16x2(float a, float b) {
  return __builtin_amdgcn_perm(rne_hi(b), rne_hi(a), 0x07060302u);
}
__device__ __forceinline__ __bf16 f2bf(float f) {
  unsigned short h = (unsigned short)(rne_hi(f) >> 16);
  return __builtin_bit_cast(__bf16, h);
}

// 16-byte global->LDS copy: async DMA path on gfx1250, sync fallback otherwise.
__device__ __forceinline__ void async_copy16(const void* g, void* l) {
#if HAVE_ASYNC
  __builtin_amdgcn_global_load_async_to_lds_b128(
      (__attribute__((address_space(1))) v4i*)(void*)g,
      (__attribute__((address_space(3))) v4i*)l, 0, 0);
#else
  *(uint4*)l = *(const uint4*)g;
#endif
}
__device__ __forceinline__ void wait_async0() {
#if HAVE_ASYNC
#if __has_builtin(__builtin_amdgcn_s_wait_asynccnt)
  __builtin_amdgcn_s_wait_asynccnt(0);
#else
  asm volatile("s_wait_asynccnt 0x0" ::: "memory");
#endif
#endif
}

// A fragment: lane group g = lane>>4; row m = lane&15;
// element i: K = ((i>>3)<<4) + g*8 + (i&7) -> two contiguous 8-elem runs.
__device__ __forceinline__ v16bf frag_a(const __bf16* rowptr) {
  int g = (threadIdx.x >> 4) & 1;
  bf16x8 lo = *(const bf16x8*)(rowptr + g * 8);
  bf16x8 hi = *(const bf16x8*)(rowptr + 16 + g * 8);
  return __builtin_shufflevector(lo, hi, 0,1,2,3,4,5,6,7,8,9,10,11,12,13,14,15);
}
// B fragment: lane holds col n=lane&15; K = ((i>>3)<<4) + g*8 + (i&7) -> two contiguous 8-elem runs (col-major tile).
__device__ __forceinline__ v16bf frag_b(const __bf16* colptr) {
  int g = (threadIdx.x >> 4) & 1;
  bf16x8 lo = *(const bf16x8*)(colptr + g * 8);
  bf16x8 hi = *(const bf16x8*)(colptr + 16 + g * 8);
  return __builtin_shufflevector(lo, hi, 0,1,2,3,4,5,6,7,8,9,10,11,12,13,14,15);
}

#define WMMA_BF16(a, b, c) \
  __builtin_amdgcn_wmma_f32_16x16x32_bf16(false, (a), false, (b), (short)0, (c), false, false)

// ---------------- small utility kernels ----------------
__global__ void zero_f32_kernel(float* __restrict__ p, int n) {
  int i = blockIdx.x * blockDim.x + threadIdx.x;
  if (i < n) p[i] = 0.0f;
}
__global__ void init_counts_kernel(int* __restrict__ p) {
  if (threadIdx.x < 16) p[threadIdx.x] = 0;  // counts[8] + cursor[8]
}

// ---------------- router: one wave32 per token ----------------
__global__ __launch_bounds__(256) void router_kernel(
    const float* __restrict__ x, const float* __restrict__ rw,
    int* __restrict__ top_e, float* __restrict__ top_w, int* __restrict__ counts) {
  int t = (blockIdx.x * blockDim.x + threadIdx.x) >> 5;
  int lane = threadIdx.x & 31;
  if (t >= T_TOK) return;
  float l[NEXP];
#pragma unroll
  for (int e = 0; e < NEXP; ++e) l[e] = 0.0f;
  const float* xr = x + (size_t)t * D_DIM;
  for (int d = lane; d < D_DIM; d += 32) {
    float xv = xr[d];
#pragma unroll
    for (int e = 0; e < NEXP; ++e) l[e] += xv * rw[e * D_DIM + d];
  }
#pragma unroll
  for (int off = 16; off > 0; off >>= 1) {
#pragma unroll
    for (int e = 0; e < NEXP; ++e) l[e] += __shfl_xor(l[e], off, 32);
  }
  if (lane == 0) {
    int e0 = 0;
#pragma unroll
    for (int e = 1; e < NEXP; ++e) if (l[e] > l[e0]) e0 = e;
    int e1 = (e0 == 0) ? 1 : 0;
#pragma unroll
    for (int e = 0; e < NEXP; ++e)
      if (e != e0 && l[e] > l[e1]) e1 = e;
    float p1 = __expf(l[e1] - l[e0]);
    float s  = 1.0f + p1;
    top_e[2 * t]     = e0;
    top_e[2 * t + 1] = e1;
    top_w[2 * t]     = 1.0f / s;
    top_w[2 * t + 1] = p1 / s;
    atomicAdd(&counts[e0], 1);
    atomicAdd(&counts[e1], 1);
  }
}

__global__ void scan_kernel(const int* __restrict__ counts,
                            int* __restrict__ offsets, int* __restrict__ cursor) {
  if (threadIdx.x == 0) {
    int acc = 0;
    for (int e = 0; e < NEXP; ++e) { offsets[e] = acc; cursor[e] = acc; acc += counts[e]; }
  }
}

__global__ void scatter_kernel(const int* __restrict__ top_e, const float* __restrict__ top_w,
                               int* __restrict__ cursor, int* __restrict__ list,
                               float* __restrict__ wlist) {
  int t = blockIdx.x * blockDim.x + threadIdx.x;
  if (t >= T_TOK) return;
#pragma unroll
  for (int k = 0; k < 2; ++k) {
    int e = top_e[2 * t + k];
    int slot = atomicAdd(&cursor[e], 1);
    list[slot]  = t;
    wlist[slot] = top_w[2 * t + k];
  }
}

// ---------------- weight convert + transpose: [K][N] f32 -> [N][K] bf16 ----------------
__global__ __launch_bounds__(256) void transpose_cvt_kernel(
    const float* __restrict__ src, __bf16* __restrict__ dst, int K, int N) {
  __shared__ __bf16 t[64 * 72];   // 72*2=144B stride, 16B-aligned
  const int e = blockIdx.z;
  const float* s = src + (size_t)e * K * N;
  __bf16* d = dst + (size_t)e * K * N;
  const int k0 = blockIdx.y * 64;
  const int n0 = blockIdx.x * 64;
  const int tid = threadIdx.x;
  const int rn = (tid & 15) * 4;
  const int rk = tid >> 4;
#pragma unroll
  for (int i = 0; i < 4; ++i) {
    int kl = rk + 16 * i;
    float4 v = *(const float4*)(s + (size_t)(k0 + kl) * N + n0 + rn);
    t[(rn + 0) * 72 + kl] = f2bf(v.x);
    t[(rn + 1) * 72 + kl] = f2bf(v.y);
    t[(rn + 2) * 72 + kl] = f2bf(v.z);
    t[(rn + 3) * 72 + kl] = f2bf(v.w);
  }
  __syncthreads();
#pragma unroll
  for (int i = 0; i < 2; ++i) {
    int ch = tid + 256 * i;
    int nl = ch >> 3;
    int kq = (ch & 7) * 8;
    bf16x8 vv = *(const bf16x8*)&t[nl * 72 + kq];
    *(bf16x8*)(d + (size_t)(n0 + nl) * K + k0 + kq) = vv;
  }
}

// ================= bf16-weight (transposed) GEMM path =================
// fc: A = gathered x (fp32->bf16 via regs), B = wfcT[e][f][k] bf16 via async copy.
__global__ __launch_bounds__(256) void moe_fc_tb_kernel(
    const float* __restrict__ x, const __bf16* __restrict__ wT, const float* __restrict__ b_fc,
    const int* __restrict__ counts, const int* __restrict__ offsets,
    const int* __restrict__ list, __bf16* __restrict__ h) {
  const int e  = blockIdx.z;
  const int by = blockIdx.y;
  const int bx = blockIdx.x;
  const int cnt = counts[e];
  if (by * BM >= cnt) return;
  const int slotBase = offsets[e] + by * BM;
  const int rem = (cnt - by * BM < BM) ? cnt - by * BM : BM;

  __shared__ __bf16 ldsA[2 * BM * LDA];
  __shared__ __bf16 ldsB[2 * BN * LDB];
  __shared__ int    srow[BM];

  const int tid = threadIdx.x;
  if (tid < BM) srow[tid] = list[slotBase + ((tid < rem) ? tid : rem - 1)];
  __syncthreads();

  const int arow = tid >> 3;
  const int acol = (tid & 7) * 4;
  const float* aptr[4];
#pragma unroll
  for (int p = 0; p < 4; ++p)
    aptr[p] = x + (size_t)srow[arow + 32 * p] * D_DIM + acol;

  // B async chunks: 512 x 16B per stage; this thread's two chunks
  const __bf16* bsrc[2];
  int bofs[2];
#pragma unroll
  for (int i = 0; i < 2; ++i) {
    int ch = tid + 256 * i;
    int col = ch >> 2;
    int kq = (ch & 3) * 8;
    bsrc[i] = wT + (size_t)e * D_DIM * F_DIM + (size_t)(bx * BN + col) * D_DIM + kq;
    bofs[i] = col * LDB + kq;
  }

  const int wave = tid >> 5;
  const int lane = tid & 31;
  const int wm = wave & 3;
  const int wn = wave >> 2;
  const int m0 = lane & 15;

  v8f acc[2][4];
#pragma unroll
  for (int af = 0; af < 2; ++af)
#pragma unroll
    for (int j = 0; j < 4; ++j)
#pragma unroll
      for (int r = 0; r < 8; ++r) acc[af][j][r] = 0.0f;

  // prologue: stage 0
  float4 aReg[4];
#pragma unroll
  for (int p = 0; p < 4; ++p) aReg[p] = *(const float4*)(aptr[p]);
#pragma unroll
  for (int i = 0; i < 2; ++i) async_copy16(bsrc[i], ldsB + bofs[i]);

  int ib = 0;
  for (int kc = 0; kc < D_DIM; kc += KS, ib ^= 1) {
    __bf16* aBase = ldsA + ib * (BM * LDA);
    const __bf16* bBase = ldsB + ib * (BN * LDB);
#pragma unroll
    for (int p = 0; p < 4; ++p) {
      uint2 pk;
      pk.x = pack_bf16x2(aReg[p].x, aReg[p].y);
      pk.y = pack_bf16x2(aReg[p].z, aReg[p].w);
      *(uint2*)&aBase[(arow + 32 * p) * LDA + acol] = pk;
    }
    wait_async0();
    __syncthreads();
    const int kn = kc + KS;
    if (kn < D_DIM) {
#pragma unroll
      for (int i = 0; i < 2; ++i)
        async_copy16(bsrc[i] + kn, ldsB + (ib ^ 1) * (BN * LDB) + bofs[i]);
#pragma unroll
      for (int p = 0; p < 4; ++p) aReg[p] = *(const float4*)(aptr[p] + kn);
    }
    v16bf a0 = frag_a(&aBase[(wm * 32 + m0) * LDA]);
    v16bf a1 = frag_a(&aBase[(wm * 32 + 16 + m0) * LDA]);
#pragma unroll
    for (int j = 0; j < 4; ++j) {
      v16bf b = frag_b(&bBase[(wn * 64 + j * 16 + m0) * LDB]);
      acc[0][j] = WMMA_BF16(a0, b, acc[0][j]);
      acc[1][j] = WMMA_BF16(a1, b, acc[1][j]);
    }
    __syncthreads();
  }

  const int g  = lane >> 4;
  const int n0 = lane & 15;
#pragma unroll
  for (int af = 0; af < 2; ++af) {
#pragma unroll
    for (int j = 0; j < 4; ++j) {
      int col = bx * BN + wn * 64 + j * 16 + n0;
      float bias = b_fc[e * F_DIM + col];
#pragma unroll
      for (int r = 0; r < 8; ++r) {
        int row = wm * 32 + af * 16 + g * 8 + r;
        if (row < rem) {
          float v = acc[af][j][r] + bias;
          float gl = 0.5f * v * (1.0f + erff(v * 0.70710678118f));
          h[(size_t)(slotBase + row) * F_DIM + col] = f2bf(gl);
        }
      }
    }
  }
}

// proj: A = h (bf16, async copy), B = wprojT[e][d][k] bf16 (async copy).
__global__ __launch_bounds__(256) void moe_proj_tb_kernel(
    const __bf16* __restrict__ h, const __bf16* __restrict__ wT,
    const float* __restrict__ b_proj,
    const int* __restrict__ counts, const int* __restrict__ offsets,
    const int* __restrict__ list, const float* __restrict__ wlist,
    float* __restrict__ out) {
  const int e  = blockIdx.z;
  const int by = blockIdx.y;
  const int bx = blockIdx.x;
  const int cnt = counts[e];
  if (by * BM >= cnt) return;
  const int slotBase = offsets[e] + by * BM;
  const int rem = (cnt - by * BM < BM) ? cnt - by * BM : BM;

  __shared__ __bf16 ldsA[2 * BM * LDA];
  __shared__ __bf16 ldsB[2 * BN * LDB];
  __shared__ int    srow[BM];
  __shared__ float  swt[BM];

  const int tid = threadIdx.x;
  if (tid < BM) {
    srow[tid] = (tid < rem) ? list[slotBase + tid] : -1;
    swt[tid]  = (tid < rem) ? wlist[slotBase + tid] : 0.0f;
  }
  __syncthreads();

  // A async chunks (rows clamped; garbage feeds only discarded rows)
  const __bf16* asrc[2];
  int aofs[2];
#pragma unroll
  for (int i = 0; i < 2; ++i) {
    int ch = tid + 256 * i;
    int row = ch >> 2;
    int c8 = (ch & 3) * 8;
    int rc = (row < rem) ? row : rem - 1;
    asrc[i] = h + (size_t)(slotBase + rc) * F_DIM + c8;
    aofs[i] = row * LDA + c8;
  }
  const __bf16* bsrc[2];
  int bofs[2];
#pragma unroll
  for (int i = 0; i < 2; ++i) {
    int ch = tid + 256 * i;
    int col = ch >> 2;
    int kq = (ch & 3) * 8;
    bsrc[i] = wT + (size_t)e * D_DIM * F_DIM + (size_t)(bx * BN + col) * F_DIM + kq;
    bofs[i] = col * LDB + kq;
  }

  const int wave = tid >> 5;
  const int lane = tid & 31;
  const int wm = wave & 3;
  const int wn = wave >> 2;
  const int m0 = lane & 15;

  v8f acc[2][4];
#pragma unroll
  for (int af = 0; af < 2; ++af)
#pragma unroll
    for (int j = 0; j < 4; ++j)
#pragma unroll
      for (int r = 0; r < 8; ++r) acc[af][j][r] = 0.0f;

  // prologue: stage 0
#pragma unroll
  for (int i = 0; i < 2; ++i) {
    async_copy16(asrc[i], ldsA + aofs[i]);
    async_copy16(bsrc[i], ldsB + bofs[i]);
  }

  int ib = 0;
  for (int kc = 0; kc < F_DIM; kc += KS, ib ^= 1) {
    const __bf16* aBase = ldsA + ib * (BM * LDA);
    const __bf16* bBase = ldsB + ib * (BN * LDB);
    wait_async0();
    __syncthreads();
    const int kn = kc + KS;
    if (kn < F_DIM) {
#pragma unroll
      for (int i = 0; i < 2; ++i) {
        async_copy16(asrc[i] + kn, ldsA + (ib ^ 1) * (BM * LDA) + aofs[i]);
        async_copy16(bsrc[i] + kn, ldsB + (ib ^ 1) * (BN * LDB) + bofs[i]);
      }
    }
    v16bf a0 = frag_a(&aBase[(wm * 32 + m0) * LDA]);
    v16bf a1 = frag_a(&aBase[(wm * 32 + 16 + m0) * LDA]);
#pragma unroll
    for (int j = 0; j < 4; ++j) {
      v16bf b = frag_b(&bBase[(wn * 64 + j * 16 + m0) * LDB]);
      acc[0][j] = WMMA_BF16(a0, b, acc[0][j]);
      acc[1][j] = WMMA_BF16(a1, b, acc[1][j]);
    }
    __syncthreads();
  }

  const int g  = lane >> 4;
  const int n0 = lane & 15;
#pragma unroll
  for (int af = 0; af < 2; ++af) {
#pragma unroll
    for (int j = 0; j < 4; ++j) {
      int col = bx * BN + wn * 64 + j * 16 + n0;
      float bias = b_proj[e * D_DIM + col];
#pragma unroll
      for (int r = 0; r < 8; ++r) {
        int row = wm * 32 + af * 16 + g * 8 + r;
        if (row < rem) {
          int tok  = srow[row];
          float v  = acc[af][j][r] + bias;
          unsafeAtomicAdd(&out[(size_t)tok * D_DIM + col], swt[row] * v);
        }
      }
    }
  }
}

// ================= fallback fp32-weight GEMM path (small ws) =================
__global__ __launch_bounds__(256) void moe_fc_kernel(
    const float* __restrict__ x, const float* __restrict__ w_fc, const float* __restrict__ b_fc,
    const int* __restrict__ counts, const int* __restrict__ offsets,
    const int* __restrict__ list, __bf16* __restrict__ h) {
  const int e  = blockIdx.z;
  const int by = blockIdx.y;
  const int bx = blockIdx.x;
  const int cnt = counts[e];
  if (by * BM >= cnt) return;
  const int slotBase = offsets[e] + by * BM;
  const int rem = (cnt - by * BM < BM) ? cnt - by * BM : BM;

  __shared__ __bf16 ldsA[BM * LDA];
  __shared__ __bf16 ldsB[BN * LDB];
  __shared__ int    srow[BM];

  const int tid = threadIdx.x;
  if (tid < BM) srow[tid] = list[slotBase + ((tid < rem) ? tid : rem - 1)];
  __syncthreads();

  const int arow = tid >> 3;
  const int acol = (tid & 7) * 4;
  const float* aptr[4];
#pragma unroll
  for (int p = 0; p < 4; ++p)
    aptr[p] = x + (size_t)srow[arow + 32 * p] * D_DIM + acol;

  const int bcol = tid & 127;
  const int bkh  = (tid >> 7) * 16;
  const float* bptr = w_fc + ((size_t)e * D_DIM + bkh) * F_DIM + (size_t)bx * BN + bcol;

  const int wave = tid >> 5;
  const int lane = tid & 31;
  const int wm = wave & 3;
  const int wn = wave >> 2;
  const int m0 = lane & 15;

  v8f acc[2][4];
#pragma unroll
  for (int af = 0; af < 2; ++af)
#pragma unroll
    for (int j = 0; j < 4; ++j)
#pragma unroll
      for (int r = 0; r < 8; ++r) acc[af][j][r] = 0.0f;

  float4 aReg[4];
  float  bReg[16];
#pragma unroll
  for (int p = 0; p < 4; ++p) aReg[p] = *(const float4*)(aptr[p]);
#pragma unroll
  for (int i = 0; i < 16; ++i) bReg[i] = bptr[(size_t)i * F_DIM];

  for (int kc = 0; kc < D_DIM; kc += KS) {
#pragma unroll
    for (int p = 0; p < 4; ++p) {
      uint2 pk;
      pk.x = pack_bf16x2(aReg[p].x, aReg[p].y);
      pk.y = pack_bf16x2(aReg[p].z, aReg[p].w);
      *(uint2*)&ldsA[(arow + 32 * p) * LDA + acol] = pk;
    }
    {
      uint4 q0, q1;
      q0.x = pack_bf16x2(bReg[0],  bReg[1]);  q0.y = pack_bf16x2(bReg[2],  bReg[3]);
      q0.z = pack_bf16x2(bReg[4],  bReg[5]);  q0.w = pack_bf16x2(bReg[6],  bReg[7]);
      q1.x = pack_bf16x2(bReg[8],  bReg[9]);  q1.y = pack_bf16x2(bReg[10], bReg[11]);
      q1.z = pack_bf16x2(bReg[12], bReg[13]); q1.w = pack_bf16x2(bReg[14], bReg[15]);
      *(uint4*)&ldsB[bcol * LDB + bkh]     = q0;
      *(uint4*)&ldsB[bcol * LDB + bkh + 8] = q1;
    }
    __syncthreads();
    const int kn = kc + KS;
    if (kn < D_DIM) {
#pragma unroll
      for (int p = 0; p < 4; ++p) aReg[p] = *(const float4*)(aptr[p] + kn);
#pragma unroll
      for (int i = 0; i < 16; ++i) bReg[i] = bptr[(size_t)kn * F_DIM + (size_t)i * F_DIM];
    }
    v16bf a0 = frag_a(&ldsA[(wm * 32 + m0) * LDA]);
    v16bf a1 = frag_a(&ldsA[(wm * 32 + 16 + m0) * LDA]);
#pragma unroll
    for (int j = 0; j < 4; ++j) {
      v16bf b = frag_b(&ldsB[(wn * 64 + j * 16 + m0) * LDB]);
      acc[0][j] = WMMA_BF16(a0, b, acc[0][j]);
      acc[1][j] = WMMA_BF16(a1, b, acc[1][j]);
    }
    __syncthreads();
  }

  const int g  = lane >> 4;
  const int n0 = lane & 15;
#pragma unroll
  for (int af = 0; af < 2; ++af) {
#pragma unroll
    for (int j = 0; j < 4; ++j) {
      int col = bx * BN + wn * 64 + j * 16 + n0;
      float bias = b_fc[e * F_DIM + col];
#pragma unroll
      for (int r = 0; r < 8; ++r) {
        int row = wm * 32 + af * 16 + g * 8 + r;
        if (row < rem) {
          float v = acc[af][j][r] + bias;
          float gl = 0.5f * v * (1.0f + erff(v * 0.70710678118f));
          h[(size_t)(slotBase + row) * F_DIM + col] = f2bf(gl);
        }
      }
    }
  }
}

__global__ __launch_bounds__(256) void moe_proj_kernel(
    const __bf16* __restrict__ h, const float* __restrict__ w_proj,
    const float* __restrict__ b_proj,
    const int* __restrict__ counts, const int* __restrict__ offsets,
    const int* __restrict__ list, const float* __restrict__ wlist,
    float* __restrict__ out) {
  const int e  = blockIdx.z;
  const int by = blockIdx.y;
  const int bx = blockIdx.x;
  const int cnt = counts[e];
  if (by * BM >= cnt) return;
  const int slotBase = offsets[e] + by * BM;
  const int rem = (cnt - by * BM < BM) ? cnt - by * BM : BM;

  __shared__ __bf16 ldsA[BM * LDA];
  __shared__ __bf16 ldsB[BN * LDB];
  __shared__ int    srow[BM];
  __shared__ float  swt[BM];

  const int tid = threadIdx.x;
  if (tid < BM) {
    srow[tid] = (tid < rem) ? list[slotBase + tid] : -1;
    swt[tid]  = (tid < rem) ? wlist[slotBase + tid] : 0.0f;
  }
  __syncthreads();

  const int arow = tid >> 2;
  const int ac8  = (tid & 3) * 8;
  const __bf16* hptr[2];
#pragma unroll
  for (int p = 0; p < 2; ++p) {
    int rr = arow + 64 * p;
    int rc = (rr < rem) ? rr : rem - 1;
    hptr[p] = h + (size_t)(slotBase + rc) * F_DIM + ac8;
  }

  const int bcol = tid & 127;
  const int bkh  = (tid >> 7) * 16;
  const float* bptr = w_proj + ((size_t)e * F_DIM + bkh) * D_DIM + (size_t)bx * BN + bcol;

  const int wave = tid >> 5;
  const int lane = tid & 31;
  const int wm = wave & 3;
  const int wn = wave >> 2;
  const int m0 = lane & 15;

  v8f acc[2][4];
#pragma unroll
  for (int af = 0; af < 2; ++af)
#pragma unroll
    for (int j = 0; j < 4; ++j)
#pragma unroll
      for (int r = 0; r < 8; ++r) acc[af][j][r] = 0.0f;

  uint4 aReg[2];
  float bReg[16];
#pragma unroll
  for (int p = 0; p < 2; ++p) aReg[p] = *(const uint4*)(hptr[p]);
#pragma unroll
  for (int i = 0; i < 16; ++i) bReg[i] = bptr[(size_t)i * D_DIM];

  for (int kc = 0; kc < F_DIM; kc += KS) {
#pragma unroll
    for (int p = 0; p < 2; ++p)
      *(uint4*)&ldsA[(arow + 64 * p) * LDA + ac8] = aReg[p];
    {
      uint4 q0, q1;
      q0.x = pack_bf16x2(bReg[0],  bReg[1]);  q0.y = pack_bf16x2(bReg[2],  bReg[3]);
      q0.z = pack_bf16x2(bReg[4],  bReg[5]);  q0.w = pack_bf16x2(bReg[6],  bReg[7]);
      q1.x = pack_bf16x2(bReg[8],  bReg[9]);  q1.y = pack_bf16x2(bReg[10], bReg[11]);
      q1.z = pack_bf16x2(bReg[12], bReg[13]); q1.w = pack_bf16x2(bReg[14], bReg[15]);
      *(uint4*)&ldsB[bcol * LDB + bkh]     = q0;
      *(uint4*)&ldsB[bcol * LDB + bkh + 8] = q1;
    }
    __syncthreads();
    const int kn = kc + KS;
    if (kn < F_DIM) {
#pragma unroll
      for (int p = 0; p < 2; ++p) aReg[p] = *(const uint4*)(hptr[p] + kn);
#pragma unroll
      for (int i = 0; i < 16; ++i) bReg[i] = bptr[(size_t)kn * D_DIM + (size_t)i * D_DIM];
    }
    v16bf a0 = frag_a(&ldsA[(wm * 32 + m0) * LDA]);
    v16bf a1 = frag_a(&ldsA[(wm * 32 + 16 + m0) * LDA]);
#pragma unroll
    for (int j = 0; j < 4; ++j) {
      v16bf b = frag_b(&ldsB[(wn * 64 + j * 16 + m0) * LDB]);
      acc[0][j] = WMMA_BF16(a0, b, acc[0][j]);
      acc[1][j] = WMMA_BF16(a1, b, acc[1][j]);
    }
    __syncthreads();
  }

  const int g  = lane >> 4;
  const int n0 = lane & 15;
#pragma unroll
  for (int af = 0; af < 2; ++af) {
#pragma unroll
    for (int j = 0; j < 4; ++j) {
      int col = bx * BN + wn * 64 + j * 16 + n0;
      float bias = b_proj[e * D_DIM + col];
#pragma unroll
      for (int r = 0; r < 8; ++r) {
        int row = wm * 32 + af * 16 + g * 8 + r;
        if (row < rem) {
          int tok  = srow[row];
          float v  = acc[af][j][r] + bias;
          unsafeAtomicAdd(&out[(size_t)tok * D_DIM + col], swt[row] * v);
        }
      }
    }
  }
}

// ---------------- host launch ----------------
extern "C" void kernel_launch(void* const* d_in, const int* in_sizes, int n_in,
                              void* d_out, int out_size, void* d_ws, size_t ws_size,
                              hipStream_t stream) {
  const float* x        = (const float*)d_in[0];  // [4,1024,768]
  const float* router_w = (const float*)d_in[1];  // [8,768]
  const float* w_fc     = (const float*)d_in[2];  // [8,768,3072]
  const float* b_fc     = (const float*)d_in[3];  // [8,3072]
  const float* w_proj   = (const float*)d_in[4];  // [8,3072,768]
  const float* b_proj   = (const float*)d_in[5];  // [8,768]
  float* out = (float*)d_out;

  char* ws = (char*)d_ws;
  int*   counts  = (int*)ws;                                 // [8]
  int*   cursor  = counts + 8;                               // [8]
  int*   offsets = cursor + 8;                               // [8]
  int*   top_e   = (int*)(ws + 128);                         // [2T]
  float* top_w   = (float*)(ws + 128 + 2 * T_TOK * 4);       // [2T]
  int*   list    = (int*)(ws + 128 + 4 * T_TOK * 4);         // [2T]
  float* wlist   = (float*)(ws + 128 + 6 * T_TOK * 4);       // [2T]

  const size_t HOFF    = 256 * 1024;
  const size_t HBYTES  = (size_t)2 * T_TOK * F_DIM * 2;          // 50.3 MB
  const size_t WBYTES  = (size_t)NEXP * D_DIM * F_DIM * 2;       // 37.7 MB each
  __bf16* hbuf   = (__bf16*)(ws + HOFF);
  __bf16* wfcT   = (__bf16*)(ws + HOFF + HBYTES);
  __bf16* wprojT = (__bf16*)(ws + HOFF + HBYTES + WBYTES);
  const bool big = ws_size >= HOFF + HBYTES + 2 * WBYTES;

  int outN = T_TOK * D_DIM;
  zero_f32_kernel<<<(outN + 255) / 256, 256, 0, stream>>>(out, outN);
  init_counts_kernel<<<1, 32, 0, stream>>>(counts);

  router_kernel<<<(T_TOK * 32) / 256, 256, 0, stream>>>(x, router_w, top_e, top_w, counts);
  scan_kernel<<<1, 32, 0, stream>>>(counts, offsets, cursor);
  scatter_kernel<<<(T_TOK + 255) / 256, 256, 0, stream>>>(top_e, top_w, cursor, list, wlist);

  dim3 gfc(F_DIM / BN, T_TOK / BM, NEXP);
  dim3 gpr(D_DIM / BN, T_TOK / BM, NEXP);

  if (big) {
    // convert+transpose weights to bf16 once per call
    dim3 gt1(F_DIM / 64, D_DIM / 64, NEXP);   // w_fc: K=768, N=3072
    transpose_cvt_kernel<<<gt1, 256, 0, stream>>>(w_fc, wfcT, D_DIM, F_DIM);
    dim3 gt2(D_DIM / 64, F_DIM / 64, NEXP);   // w_proj: K=3072, N=768
    transpose_cvt_kernel<<<gt2, 256, 0, stream>>>(w_proj, wprojT, F_DIM, D_DIM);

    moe_fc_tb_kernel<<<gfc, 256, 0, stream>>>(x, wfcT, b_fc, counts, offsets, list, hbuf);
    moe_proj_tb_kernel<<<gpr, 256, 0, stream>>>(hbuf, wprojT, b_proj, counts, offsets, list,
                                                wlist, out);
  } else {
    moe_fc_kernel<<<gfc, 256, 0, stream>>>(x, w_fc, b_fc, counts, offsets, list, hbuf);
    moe_proj_kernel<<<gpr, 256, 0, stream>>>(hbuf, w_proj, b_proj, counts, offsets, list,
                                             wlist, out);
  }
}
